// MM_Model_Simple_Enhanced_15710990369585
// MI455X (gfx1250) — matrix operand order repaired and checked by reading, not compiled, not run on hardware
//
#include <hip/hip_runtime.h>
#include <math.h>

typedef __attribute__((ext_vector_type(16))) __bf16 v16bf;
typedef __attribute__((ext_vector_type(8)))  __bf16 v8bf;
typedef __attribute__((ext_vector_type(8)))  float  v8f;

#define BM 128
#define BN 128
#define BK 32
#define BKP 40   // +8 bf16 pad: row stride 80B keeps 16B alignment, spreads banks

// ---------------------------------------------------------------------------
// Generic GEMM: C[M,N] = act(A[M,K](f32) @ B[K,N](f32) + bias[N]),
// f32 inputs converted to bf16 on load, f32 WMMA accumulation.
// Software-pipelined: register prefetch + ping-pong LDS, one barrier per step.
// Requires M%128==0, N%128==0, K%32==0. ACT: 0=none, 1=relu.
// ---------------------------------------------------------------------------
template <int ACT>
__global__ __launch_bounds__(256)
void gemm_wmma_bf16(const float* __restrict__ A, const float* __restrict__ B,
                    const float* __restrict__ bias, float* __restrict__ C,
                    int M, int N, int K, int lda, int ldb, int ldc) {
  __shared__ __bf16 As[2][BM][BKP];   // row-major M x K tile (ping-pong)
  __shared__ __bf16 Bs[2][BN][BKP];   // B tile stored TRANSPOSED: [n][k]

  const int m0   = blockIdx.x * BM;
  const int n0   = blockIdx.y * BN;
  const int tid  = threadIdx.x;
  const int lane = tid & 31;
  const int wid  = tid >> 5;
  const int wm   = (wid & 3) * 32;  // 4 waves along M, 32 rows each
  const int wn   = (wid >> 2) * 64; // 2 waves along N, 64 cols each

  v8f acc[2][4];
#pragma unroll
  for (int i = 0; i < 2; ++i)
#pragma unroll
    for (int j = 0; j < 4; ++j)
      acc[i][j] = (v8f){0.f, 0.f, 0.f, 0.f, 0.f, 0.f, 0.f, 0.f};

  const int arow = tid >> 1;          // 0..127
  const int acol = (tid & 1) * 16;    // 0 or 16
  const int bn   = tid & 127;         // 0..127
  const int bk0  = (tid >> 7) * 16;   // 0 or 16

  // register staging for the next tile (f32)
  float4 aR[4];
  float  bR[16];

  auto loadTiles = [&](int k0) {
    const float4* ap =
        reinterpret_cast<const float4*>(A + (size_t)(m0 + arow) * lda + k0 + acol);
#pragma unroll
    for (int c = 0; c < 4; ++c) aR[c] = ap[c];
#pragma unroll
    for (int i = 0; i < 16; ++i)
      bR[i] = B[(size_t)(k0 + bk0 + i) * ldb + n0 + bn];
  };
  auto stageTiles = [&](int buf) {
#pragma unroll
    for (int c = 0; c < 4; ++c) {
      As[buf][arow][acol + 4 * c + 0] = (__bf16)aR[c].x;
      As[buf][arow][acol + 4 * c + 1] = (__bf16)aR[c].y;
      As[buf][arow][acol + 4 * c + 2] = (__bf16)aR[c].z;
      As[buf][arow][acol + 4 * c + 3] = (__bf16)aR[c].w;
    }
#pragma unroll
    for (int i = 0; i < 16; ++i) Bs[buf][bn][bk0 + i] = (__bf16)bR[i];
  };

  loadTiles(0);
  stageTiles(0);
  __syncthreads();

  int buf = 0;
  for (int k0 = 0; k0 < K; k0 += BK) {
    const bool hasNext = (k0 + BK) < K;
    if (hasNext) loadTiles(k0 + BK);  // global loads overlap WMMA below

    // A fragment: lane<16 holds row, K {kb..kb+7, kb+16..kb+23}, kb=(lane>>4)*8
    v16bf afrag[2];
#pragma unroll
    for (int i = 0; i < 2; ++i) {
      int r  = wm + i * 16 + (lane & 15);
      int kb = (lane >> 4) * 8;
      v8bf lo = *(const v8bf*)&As[buf][r][kb];
      v8bf hi = *(const v8bf*)&As[buf][r][kb + 16];
      afrag[i] = __builtin_shufflevector(lo, hi, 0, 1, 2, 3, 4, 5, 6, 7,
                                         8, 9, 10, 11, 12, 13, 14, 15);
    }
    // B fragment: lane holds column, K {kb..kb+7, kb+16..kb+23}, kb=(lane>>4)*8
    v16bf bfrag[4];
#pragma unroll
    for (int j = 0; j < 4; ++j) {
      int cc = wn + j * 16 + (lane & 15);
      int kb = (lane >> 4) * 8;
      v8bf lo = *(const v8bf*)&Bs[buf][cc][kb];
      v8bf hi = *(const v8bf*)&Bs[buf][cc][kb + 16];
      bfrag[j] = __builtin_shufflevector(lo, hi, 0, 1, 2, 3, 4, 5, 6, 7,
                                         8, 9, 10, 11, 12, 13, 14, 15);
    }
#pragma unroll
    for (int i = 0; i < 2; ++i)
#pragma unroll
      for (int j = 0; j < 4; ++j)
        acc[i][j] = __builtin_amdgcn_wmma_f32_16x16x32_bf16(
            false, afrag[i], false, bfrag[j], (short)0, acc[i][j], false, false);

    if (hasNext) {
      stageTiles(buf ^ 1);   // write the other buffer: no race with reads above
      __syncthreads();
      buf ^= 1;
    }
  }

  // epilogue: C/D layout — VGPR r holds M = r + (lane>>4)*8, N = lane&15
#pragma unroll
  for (int i = 0; i < 2; ++i) {
#pragma unroll
    for (int j = 0; j < 4; ++j) {
      int col  = n0 + wn + j * 16 + (lane & 15);
      float bv = bias ? bias[col] : 0.f;
#pragma unroll
      for (int r = 0; r < 8; ++r) {
        int row = m0 + wm + i * 16 + (lane >> 4) * 8 + r;
        float v = acc[i][j][r] + bv;
        if (ACT == 1) v = v > 0.f ? v : 0.f;
        C[(size_t)row * ldc + col] = v;
      }
    }
  }
}

// ---------------------------------------------------------------------------
// Per-row attention gate: e = x * sigmoid(relu(x@w1+b1)@w2+b2), x in R^128.
// One 64-thread block per row; writes into Ecat[row, colOff..colOff+127].
// ---------------------------------------------------------------------------
__global__ __launch_bounds__(64)
void attn_gate(const float* __restrict__ X, const float* __restrict__ w1,
               const float* __restrict__ b1, const float* __restrict__ w2,
               const float* __restrict__ b2, float* __restrict__ E,
               int ldE, int colOff) {
  __shared__ float xs[128];
  __shared__ float red[64];
  const int row = blockIdx.x;
  const int t   = threadIdx.x;
  xs[t]      = X[(size_t)row * 128 + t];
  xs[t + 64] = X[(size_t)row * 128 + 64 + t];
  __syncthreads();
  float h = b1[t];
#pragma unroll 4
  for (int k = 0; k < 128; ++k) h += xs[k] * w1[k * 64 + t];
  h = fmaxf(h, 0.f);
  red[t] = h * w2[t];
  __syncthreads();
  for (int s = 32; s > 0; s >>= 1) {
    if (t < s) red[t] += red[t + s];
    __syncthreads();
  }
  float a = 1.f / (1.f + __expf(-(red[0] + b2[0])));
  E[(size_t)row * ldE + colOff + t]      = xs[t] * a;
  E[(size_t)row * ldE + colOff + t + 64] = xs[t + 64] * a;
}

// ---------------------------------------------------------------------------
// Row-wise softmax over 128 columns (compact ld=128 in and out).
// ---------------------------------------------------------------------------
__global__ __launch_bounds__(128)
void softmax128(const float* __restrict__ in, float* __restrict__ out) {
  __shared__ float s[128];
  const int row = blockIdx.x;
  const int t   = threadIdx.x;
  float v = in[(size_t)row * 128 + t];
  s[t] = v;
  __syncthreads();
  for (int st = 64; st > 0; st >>= 1) {
    if (t < st) s[t] = fmaxf(s[t], s[t + st]);
    __syncthreads();
  }
  float m = s[0];
  __syncthreads();
  float e = __expf(v - m);
  s[t] = e;
  __syncthreads();
  for (int st = 64; st > 0; st >>= 1) {
    if (t < st) s[t] += s[t + st];
    __syncthreads();
  }
  out[(size_t)row * 128 + t] = e / s[0];
}

// Ecat[:, 0:128] += 0.1*inter ; Ecat[:, 128:256] += 0.1*inter
__global__ void addscale_dual(float* __restrict__ Ecat,
                              const float* __restrict__ inter, int n) {
  int i = blockIdx.x * blockDim.x + threadIdx.x;
  if (i >= n) return;
  int r = i >> 7, c = i & 127;
  float v = 0.1f * inter[i];
  Ecat[(size_t)r * 256 + c]       += v;
  Ecat[(size_t)r * 256 + 128 + c] += v;
}

// Rb[:,0:128]=e_img+0.1*fus ; Rb[:,128:256]=e_txt+0.1*fus ; Rb[:,384:512]=i0
__global__ void finalize_rb(const float* __restrict__ Ecat,
                            const float* __restrict__ fus,
                            const float* __restrict__ iid,
                            float* __restrict__ Rb, int n) {
  int i = blockIdx.x * blockDim.x + threadIdx.x;
  if (i >= n) return;
  int r = i >> 7, c = i & 127;
  float f = 0.1f * fus[i];
  size_t ro = (size_t)r * 640;
  Rb[ro + c]       = Ecat[(size_t)r * 256 + c] + f;
  Rb[ro + 128 + c] = Ecat[(size_t)r * 256 + 128 + c] + f;
  Rb[ro + 384 + c] = iid[i];
}

// out_u = [mean(u0,u1,u2) | user_prof | image_user | text_user | title_user]
__global__ void assemble_u(const float* __restrict__ uid,
                           const float* __restrict__ Ub,
                           const float* __restrict__ U2,
                           float* __restrict__ out, int n) {
  int i = blockIdx.x * blockDim.x + threadIdx.x;
  if (i >= n) return;
  int r = i >> 7, c = i & 127;
  size_t ro = (size_t)r * 640;
  out[ro + c]       = (uid[i] + Ub[ro + 384 + c] + U2[i]) * (1.f / 3.f);
  out[ro + 128 + c] = Ub[ro + 512 + c];   // user_prof
  out[ro + 256 + c] = Ub[ro + c];         // image_user
  out[ro + 384 + c] = Ub[ro + 128 + c];   // text_user
  out[ro + 512 + c] = Ub[ro + 256 + c];   // title_user
}

// out_i = [mean(i0,i1,i2) | item_prof | image_item | text_item | title_item]
__global__ void assemble_i(const float* __restrict__ iid,
                           const float* __restrict__ Uc,
                           const float* __restrict__ Rb,
                           const float* __restrict__ I2,
                           float* __restrict__ out, int n) {
  int i = blockIdx.x * blockDim.x + threadIdx.x;
  if (i >= n) return;
  int r = i >> 7, c = i & 127;
  size_t ro  = (size_t)r * 640;
  size_t rc  = (size_t)r * 512;
  out[ro + c]       = (iid[i] + Uc[rc + 384 + c] + I2[i]) * (1.f / 3.f);
  out[ro + 128 + c] = Rb[ro + 512 + c];   // item_prof
  out[ro + 256 + c] = Uc[rc + c];         // image_item
  out[ro + 384 + c] = Uc[rc + 128 + c];   // text_item
  out[ro + 512 + c] = Uc[rc + 256 + c];   // title_item
}

// ---------------------------------------------------------------------------
extern "C" void kernel_launch(void* const* d_in, const int* in_sizes, int n_in,
                              void* d_out, int out_size, void* d_ws,
                              size_t ws_size, hipStream_t stream) {
  (void)in_sizes; (void)n_in; (void)out_size; (void)ws_size;
  constexpr int NU = 8192, NI = 4096;

  const float* ui   = (const float*)d_in[0];
  const float* iu   = (const float*)d_in[1];
  const float* imgF = (const float*)d_in[6];
  const float* txtF = (const float*)d_in[7];
  const float* usrF = (const float*)d_in[8];
  const float* ttlF = (const float*)d_in[9];
  const float* W_img = (const float*)d_in[10]; const float* b_img = (const float*)d_in[11];
  const float* W_txt = (const float*)d_in[12]; const float* b_txt = (const float*)d_in[13];
  const float* W_usr = (const float*)d_in[14]; const float* b_usr = (const float*)d_in[15];
  const float* W_ttl = (const float*)d_in[16]; const float* b_ttl = (const float*)d_in[17];
  const float* uid  = (const float*)d_in[18];
  const float* iid  = (const float*)d_in[19];
  const float* aiw1 = (const float*)d_in[20]; const float* aib1 = (const float*)d_in[21];
  const float* aiw2 = (const float*)d_in[22]; const float* aib2 = (const float*)d_in[23];
  const float* atw1 = (const float*)d_in[24]; const float* atb1 = (const float*)d_in[25];
  const float* atw2 = (const float*)d_in[26]; const float* atb2 = (const float*)d_in[27];
  const float* fiw1 = (const float*)d_in[28]; const float* fib1 = (const float*)d_in[29];
  const float* fiw2 = (const float*)d_in[30]; const float* fib2 = (const float*)d_in[31];
  const float* efw1 = (const float*)d_in[32]; const float* efb1 = (const float*)d_in[33];
  const float* efw2 = (const float*)d_in[34]; const float* efb2 = (const float*)d_in[35];
  float* out = (float*)d_out;

  // workspace layout (floats), with lifetime-safe aliasing
  float* ws = (float*)d_ws;
  size_t off = 0;
  auto alloc = [&](size_t n) { float* p = ws + off; off += n; return p; };
  float* P_img = alloc((size_t)NI * 128);   // dead after gating
  float* P_txt = alloc((size_t)NI * 128);   // dead after gating
  float* P_usr = alloc((size_t)NU * 128);   // dead after pass A
  float* Ecat  = alloc((size_t)NI * 256);
  float* Hbuf  = alloc((size_t)NI * 128);   // dead after fusion GEMMs
  float* inter = alloc((size_t)NI * 128);   // dead after addscale
  float* fus   = alloc((size_t)NI * 128);
  float* Rb    = alloc((size_t)NI * 640);   // pass-B RHS block
  float* Ub    = alloc((size_t)NU * 640);   // pass-B output block
  float* Uc    = alloc((size_t)NI * 512);   // pass-C output block
  float* T1 = P_img;                        // [NU,128] alias (P_img+P_txt)
  float* U2 = P_usr;                        // [NU,128] alias
  float* T2 = Hbuf;                         // [NI,128] alias
  float* I2 = inter;                        // [NI,128] alias

  auto G = [](int M, int N) { return dim3((unsigned)(M / BM), (unsigned)(N / BN)); };
  const int EW = 256;                                 // elementwise block
  const int nI = NI * 128, nU = NU * 128;

  // 1) modality projections (ttl written straight into Rb cols 256:384)
  gemm_wmma_bf16<0><<<G(NI, 128), 256, 0, stream>>>(imgF, W_img, b_img, P_img, NI, 128, 4096, 4096, 128, 128);
  gemm_wmma_bf16<0><<<G(NI, 128), 256, 0, stream>>>(txtF, W_txt, b_txt, P_txt, NI, 128, 768, 768, 128, 128);
  gemm_wmma_bf16<0><<<G(NU, 128), 256, 0, stream>>>(usrF, W_usr, b_usr, P_usr, NU, 128, 768, 768, 128, 128);
  gemm_wmma_bf16<0><<<G(NI, 128), 256, 0, stream>>>(ttlF, W_ttl, b_ttl, Rb + 256, NI, 128, 768, 768, 128, 640);

  // 2) attention gating -> Ecat = [e_img | e_txt]
  attn_gate<<<NI, 64, 0, stream>>>(P_img, aiw1, aib1, aiw2, aib2, Ecat, 256, 0);
  attn_gate<<<NI, 64, 0, stream>>>(P_txt, atw1, atb1, atw2, atb2, Ecat, 256, 128);

  // 3) feature interaction + enhanced fusion
  gemm_wmma_bf16<1><<<G(NI, 128), 256, 0, stream>>>(Ecat, fiw1, fib1, Hbuf, NI, 128, 256, 256, 128, 128);
  gemm_wmma_bf16<0><<<G(NI, 128), 256, 0, stream>>>(Hbuf, fiw2, fib2, inter, NI, 128, 128, 128, 128, 128);
  addscale_dual<<<(nI + EW - 1) / EW, EW, 0, stream>>>(Ecat, inter, nI);
  gemm_wmma_bf16<1><<<G(NI, 128), 256, 0, stream>>>(Ecat, efw1, efb1, Hbuf, NI, 128, 256, 256, 128, 128);
  gemm_wmma_bf16<0><<<G(NI, 128), 256, 0, stream>>>(Hbuf, efw2, efb2, fus, NI, 128, 128, 128, 128, 128);
  finalize_rb<<<(nI + EW - 1) / EW, EW, 0, stream>>>(Ecat, fus, iid, Rb, nI);

  // 4) pass A: item_prof = iu @ usr  -> Rb cols 512:640
  gemm_wmma_bf16<0><<<G(NI, 128), 256, 0, stream>>>(iu, P_usr, nullptr, Rb + 512, NI, 128, NU, NU, 128, 640);
  // 5) pass B: [image_user|text_user|title_user|u1|user_prof] = ui @ Rb
  gemm_wmma_bf16<0><<<G(NU, 640), 256, 0, stream>>>(ui, Rb, nullptr, Ub, NU, 640, NI, NI, 640, 640);
  // 6) pass C: [image_item|text_item|title_item|i1] = iu @ Ub[:, :512]
  gemm_wmma_bf16<0><<<G(NI, 512), 256, 0, stream>>>(iu, Ub, nullptr, Uc, NI, 512, NU, NU, 640, 512);
  // 7) pass D: u2 = softmax(ui @ i1)
  gemm_wmma_bf16<0><<<G(NU, 128), 256, 0, stream>>>(ui, Uc + 384, nullptr, T1, NU, 128, NI, NI, 512, 128);
  softmax128<<<NU, 128, 0, stream>>>(T1, U2);
  // 8) pass E: i2 = softmax(iu @ u2)
  gemm_wmma_bf16<0><<<G(NI, 128), 256, 0, stream>>>(iu, U2, nullptr, T2, NI, 128, NU, NU, 128, 128);
  softmax128<<<NI, 128, 0, stream>>>(T2, I2);

  // 9) assemble final [NU+NI, 640] output
  assemble_u<<<(nU + EW - 1) / EW, EW, 0, stream>>>(uid, Ub, U2, out, nU);
  assemble_i<<<(nI + EW - 1) / EW, EW, 0, stream>>>(iid, Uc, Rb, I2, out + (size_t)NU * 640, nI);
}
